// VectorQuantizer_14645838479604
// MI455X (gfx1250) — matrix operand order repaired and checked by reading, not compiled, not run on hardware
//
#include <hip/hip_runtime.h>
#include <hip/hip_bf16.h>

// ---------------------------------------------------------------------------
// VQ forward for MI455X (gfx1250, wave32, WMMA + async global->LDS).
//  prep  : one-time codebook transform -> d_ws as 8 chunk blocks:
//            block c (20480 B): [0,512) ||e||^2 f32[128], [2048,18432) E^T f16[64][128]
//  main  : scores = X @ E^T via v_wmma_f32_16x16x32_f16, chunks double-buffered
//          in LDS via GLOBAL_LOAD_ASYNC_TO_LDS_B128 (5 ops/thread/chunk,
//          s_wait_asynccnt 0x5 => previous chunk complete, overlap with WMMA).
//          dist = ||e||^2 - 2*dot (||x||^2 dropped), argmin, f32 gather, loss.
// ---------------------------------------------------------------------------

#define N_ROWS       65536
#define DIM          64
#define NCODES       1024
#define CHUNK        128
#define NCHUNKS      (NCODES / CHUNK)          // 8
#define WAVES_PER_WG 8
#define ROWS_PER_WG  (WAVES_PER_WG * 16)       // 128
#define THREADS_WG   256
#define BLOCK_BYTES  20480                     // 256 threads * 5 * 16B
#define NORM_OFF     0                         // f32[128]
#define ET_OFF       2048                      // f16[64][128]
#define WS_BLOCKS_OFF 4096                     // blocks start here in d_ws

typedef __attribute__((ext_vector_type(16))) _Float16 v16h;
typedef __attribute__((ext_vector_type(8)))  float    v8f;
typedef __attribute__((ext_vector_type(4)))  int      v4i;

#if defined(__has_builtin)
#  if __has_builtin(__builtin_amdgcn_global_load_async_to_lds_b128)
#    define VQ_ASYNC 1
#  endif
#endif
#ifndef VQ_ASYNC
#  define VQ_ASYNC 0
#endif

#if VQ_ASYNC
#define VQ_WAIT_ASYNC(imm) asm volatile("s_wait_asynccnt " #imm ::: "memory")
#else
#define VQ_WAIT_ASYNC(imm)
#endif

// Copy one 20480B chunk block (global -> LDS); exactly 5 x B128 per thread.
__device__ __forceinline__ void chunk_copy(const unsigned char* __restrict__ g,
                                           unsigned char* l, int tid) {
    const unsigned char* gs = g + tid * 16;
    unsigned char*       ls = l + tid * 16;
#if VQ_ASYNC
    // Builtin signature (from compiler diagnostic): (v4i AS1*, v4i AS3*, Ii offset, Ii cpol)
    typedef __attribute__((address_space(1))) v4i* gp_t;
    typedef __attribute__((address_space(3))) v4i* lp_t;
    gp_t gp = (gp_t)gs;
    lp_t lp = (lp_t)ls;
    __builtin_amdgcn_global_load_async_to_lds_b128(gp, lp, 0,     0);
    __builtin_amdgcn_global_load_async_to_lds_b128(gp, lp, 4096,  0);
    __builtin_amdgcn_global_load_async_to_lds_b128(gp, lp, 8192,  0);
    __builtin_amdgcn_global_load_async_to_lds_b128(gp, lp, 12288, 0);
    __builtin_amdgcn_global_load_async_to_lds_b128(gp, lp, 16384, 0);
#else
#pragma unroll
    for (int j = 0; j < 5; ++j)
        *(int4*)(ls + j * 4096) = *(const int4*)(gs + j * 4096);
#endif
}

__global__ void vq_init(float* acc) {
    if (threadIdx.x == 0 && blockIdx.x == 0) acc[0] = 0.0f;
}

__global__ void vq_finalize(const float* __restrict__ acc, float* __restrict__ loss_out) {
    if (threadIdx.x == 0 && blockIdx.x == 0)
        loss_out[0] = 1.25f * acc[0] / (float)((size_t)N_ROWS * DIM);
}

// One-time: norms + f16 transposed codebook, chunk-blocked into d_ws.
__global__ __launch_bounds__(256)
void vq_prep(const float* __restrict__ E, unsigned char* __restrict__ blocks) {
    const int k = blockIdx.x * 256 + threadIdx.x;          // code id 0..1023
    if (k >= NCODES) return;
    const int chunk = k >> 7, col = k & (CHUNK - 1);
    unsigned char* blk = blocks + (size_t)chunk * BLOCK_BYTES;
    float*    nrm = (float*)(blk + NORM_OFF);
    _Float16* et  = (_Float16*)(blk + ET_OFF);
    const float* er = E + (size_t)k * DIM;
    float s = 0.0f;
#pragma unroll
    for (int d = 0; d < DIM; ++d) {
        float v = er[d];
        s += v * v;
        et[d * CHUNK + col] = (_Float16)v;                 // E^T[d][col]
    }
    nrm[col] = s;
}

__global__ __launch_bounds__(THREADS_WG)
void vq_main(const float* __restrict__ X,                  // [N_ROWS, DIM] f32
             const float* __restrict__ E,                  // [NCODES, DIM] f32
             const unsigned char* __restrict__ blocks,     // prepped chunk blocks
             float* __restrict__ out_q,
             float* __restrict__ out_idx,
             float* __restrict__ loss_acc)
{
    __shared__ __align__(128) unsigned char sBuf[2][BLOCK_BYTES];   // 40 KB
    __shared__ int sIdx[ROWS_PER_WG];

    const int tid  = threadIdx.x;
    const int lane = tid & 31;
    const int wave = tid >> 5;
    const int row_base = blockIdx.x * ROWS_PER_WG;
    const int wrow     = row_base + wave * 16;

    // A fragments (16-bit A 16x32 layout): lane l<16: row=l, K {0..7,16..23};
    // lane l+16: row=l, K {8..15,24..31}; a0 = D[0,32), a1 = D[32,64)
    const int arow = lane & 15;
    const int koff = (lane < 16) ? 0 : 8;
    const float* xr = X + (size_t)(wrow + arow) * DIM;
    v16h a0, a1;
#pragma unroll
    for (int j = 0; j < 8; ++j) {
        a0[j]     = (_Float16)xr[koff + j];
        a0[j + 8] = (_Float16)xr[koff + 16 + j];
        a1[j]     = (_Float16)xr[32 + koff + j];
        a1[j + 8] = (_Float16)xr[32 + koff + 16 + j];
    }

    float bestD[8];
    int   bestI[8];
#pragma unroll
    for (int i = 0; i < 8; ++i) { bestD[i] = 3.4e38f; bestI[i] = 0; }

    // prologue: start chunk 0 into buffer 0
    chunk_copy(blocks, sBuf[0], tid);

    for (int ci = 0; ci < NCHUNKS; ++ci) {
        if (ci + 1 < NCHUNKS) {
            // prefetch next chunk, then wait until <=5 outstanding:
            // in-order completion => chunk ci has fully landed in LDS
            chunk_copy(blocks + (size_t)(ci + 1) * BLOCK_BYTES, sBuf[(ci + 1) & 1], tid);
            VQ_WAIT_ASYNC(0x5);
        } else {
            VQ_WAIT_ASYNC(0x0);
        }
        __syncthreads();

        const float*    sN = (const float*)   (sBuf[ci & 1] + NORM_OFF);
        const _Float16* sE = (const _Float16*)(sBuf[ci & 1] + ET_OFF);
        const int cbase = ci * CHUNK;

#pragma unroll
        for (int sub = 0; sub < CHUNK / 16; ++sub) {
            v16h b0, b1;   // col = lane&15, K in same order as A (koff+j, koff+16+j; +32 for b1)
            const _Float16* sc = sE + sub * 16 + (lane & 15);
#pragma unroll
            for (int j = 0; j < 8; ++j) {
                b0[j]     = sc[(koff + j) * CHUNK];
                b0[j + 8] = sc[(koff + 16 + j) * CHUNK];
                b1[j]     = sc[(32 + koff + j) * CHUNK];
                b1[j + 8] = sc[(32 + koff + 16 + j) * CHUNK];
            }
            v8f c = {};
            c = __builtin_amdgcn_wmma_f32_16x16x32_f16(false, a0, false, b0,
                                                       (short)0, c, false, false);
            c = __builtin_amdgcn_wmma_f32_16x16x32_f16(false, a1, false, b1,
                                                       (short)0, c, false, false);
            const int   code = cbase + sub * 16 + (lane & 15);
            const float nrm  = sN[sub * 16 + (lane & 15)];
#pragma unroll
            for (int i = 0; i < 8; ++i) {
                float dist = nrm - 2.0f * c[i];
                if (dist < bestD[i]) { bestD[i] = dist; bestI[i] = code; }
            }
        }
        __syncthreads();   // all waves done with sBuf[ci&1] before it is refilled
    }

    // argmin reduction across the 16 lanes sharing each row (tie: lowest index)
#pragma unroll
    for (int m = 1; m < 16; m <<= 1) {
#pragma unroll
        for (int i = 0; i < 8; ++i) {
            float od = __shfl_xor(bestD[i], m, 32);
            int   oi = __shfl_xor(bestI[i], m, 32);
            if (od < bestD[i] || (od == bestD[i] && oi < bestI[i])) {
                bestD[i] = od; bestI[i] = oi;
            }
        }
    }
    if ((lane & 15) == 0) {
        int mrow = wave * 16 + ((lane < 16) ? 0 : 8);
#pragma unroll
        for (int i = 0; i < 8; ++i) sIdx[mrow + i] = bestI[i];
    }
    __syncthreads();

    // epilogue: f32 gather of chosen codes, quantized output, loss partial
    float lsum = 0.0f;
    for (int e = tid; e < ROWS_PER_WG * DIM; e += THREADS_WG) {
        const int r = e >> 6;
        const int d = e & 63;
        const int k = sIdx[r];
        const float q  = E[(size_t)k * DIM + d];
        const float x  = X[(size_t)(row_base + r) * DIM + d];
        const float df = q - x;
        lsum += df * df;
        out_q[(size_t)(row_base + r) * DIM + d] = q;
    }
    if (tid < ROWS_PER_WG) out_idx[row_base + tid] = (float)sIdx[tid];

#pragma unroll
    for (int m = 1; m < 32; m <<= 1) lsum += __shfl_xor(lsum, m, 32);
    if (lane == 0) atomicAdd(loss_acc, lsum);
}

extern "C" void kernel_launch(void* const* d_in, const int* in_sizes, int n_in,
                              void* d_out, int out_size, void* d_ws, size_t ws_size,
                              hipStream_t stream) {
    (void)in_sizes; (void)n_in; (void)out_size; (void)ws_size;
    const float* X = (const float*)d_in[0];
    const float* E = (const float*)d_in[1];
    float* out     = (float*)d_out;
    float* out_q   = out;
    float* loss    = out + (size_t)N_ROWS * DIM;
    float* out_idx = loss + 1;

    float*         acc    = (float*)d_ws;
    unsigned char* blocks = (unsigned char*)d_ws + WS_BLOCKS_OFF;

    vq_init<<<1, 1, 0, stream>>>(acc);
    vq_prep<<<(NCODES + 255) / 256, 256, 0, stream>>>(E, blocks);
    vq_main<<<N_ROWS / ROWS_PER_WG, THREADS_WG, 0, stream>>>(X, E, blocks,
                                                             out_q, out_idx, acc);
    vq_finalize<<<1, 1, 0, stream>>>(acc, loss);
}
